// MoELayer_2551210573879
// MI455X (gfx1250) — matrix operand order repaired and checked by reading, not compiled, not run on hardware
//
#include <hip/hip_runtime.h>
#include <hip/hip_bf16.h>

// ---------------------------------------------------------------------------
// MoE layer for gfx1250 (MI455X), bf16 WMMA path + async-to-LDS staging.
//   B=4, S=2048 -> NT=8192 tokens, D=1024, DFF=4096, E=8, TOPK=2
// Stages:
//   0) memset counters + output
//   1) transpose-convert W1,W2 fp32 -> bf16 (W1T [E][DFF][D], W2T [E][D][DFF])
//   2) router: logits, top-2, softmax, expert scatter lists, x -> bf16
//   3) fused grouped FFN: gelu(x@W1+b1)@W2+b2, gate-weighted atomic combine
// ---------------------------------------------------------------------------

#define NT   8192      // tokens
#define DIM  1024      // D
#define DFF_ 4096      // DFF
#define NE   8         // experts
#define MT   32        // token tile per FFN block
#define TILES_PER_E (NT / MT)   // 256 (worst case: all tokens -> one expert)

typedef __bf16 bf16x16 __attribute__((ext_vector_type(16)));
typedef float  f32x8   __attribute__((ext_vector_type(8)));

union FragU { bf16x16 v; unsigned short u[16]; };

// ---- CDNA5 async global->LDS copies (ASYNCcnt), guarded for portability ----
#ifndef __has_builtin
#define __has_builtin(x) 0
#endif
#if __has_builtin(__builtin_amdgcn_global_load_async_to_lds_b128)
#define HAVE_ASYNC_LDS 1
#else
#define HAVE_ASYNC_LDS 0
#endif

// Probe-derived signature: (AS1 int4* gaddr, AS3 int4* lds, imm offset, imm cpol)
typedef int v4i_raw __attribute__((vector_size(16)));
typedef __attribute__((address_space(1))) v4i_raw* as1_v4p;
typedef __attribute__((address_space(3))) v4i_raw* as3_v4p;

static __device__ __forceinline__ void copy16_g2lds(const void* g, void* l) {
#if HAVE_ASYNC_LDS
    // AS3 value = low 32 bits of the generic LDS address; AS1 = generic addr.
    __builtin_amdgcn_global_load_async_to_lds_b128(
        (as1_v4p)(unsigned long long)g,
        (as3_v4p)(unsigned int)(unsigned long long)l, 0, 0);
#else
    *(uint4*)l = *(const uint4*)g;
#endif
}

static __device__ __forceinline__ void async_fence() {
#if HAVE_ASYNC_LDS
#if __has_builtin(__builtin_amdgcn_s_wait_asynccnt)
    __builtin_amdgcn_s_wait_asynccnt(0);
#else
    asm volatile("s_wait_asynccnt 0x0" ::: "memory");
#endif
#endif
}

static __device__ __forceinline__ unsigned short f32_to_bf16_rn(float f) {
    union { float f; unsigned u; } c; c.f = f;
    unsigned r = c.u + 0x7fffu + ((c.u >> 16) & 1u);   // round-to-nearest-even
    return (unsigned short)(r >> 16);
}

static __device__ __forceinline__ float gelu_tanh(float x) {
    // jax.nn.gelu default (approximate=True)
    float x3 = x * x * x;
    return 0.5f * x * (1.0f + tanhf(0.7978845608028654f * (x + 0.044715f * x3)));
}

// ---------------------------------------------------------------------------
// Stage 1: tiled transpose + fp32->bf16 convert.
//   src: [E][R][C] fp32, dst: [E][C][R] bf16.  grid = (C/32, R/32, E)
// ---------------------------------------------------------------------------
__global__ __launch_bounds__(256) void transpose_cvt_kernel(
    const float* __restrict__ src, unsigned short* __restrict__ dst,
    int R, int C) {
    __shared__ float tile[32][33];
    int t  = threadIdx.x;
    int bx = blockIdx.x * 32;   // C base
    int by = blockIdx.y * 32;   // R base
    size_t base = (size_t)blockIdx.z * R * C;

    int r  = t >> 3;            // 0..31
    int c0 = (t & 7) * 4;       // 0..28
    float4 v = *(const float4*)(src + base + (size_t)(by + r) * C + bx + c0);
    tile[r][c0 + 0] = v.x; tile[r][c0 + 1] = v.y;
    tile[r][c0 + 2] = v.z; tile[r][c0 + 3] = v.w;
    __syncthreads();

    // write transposed: dst[bx + cr][by + cc] = tile[cc][cr]
    int cr = t >> 3;
    ushort4 o;
    o.x = f32_to_bf16_rn(tile[c0 + 0][cr]);
    o.y = f32_to_bf16_rn(tile[c0 + 1][cr]);
    o.z = f32_to_bf16_rn(tile[c0 + 2][cr]);
    o.w = f32_to_bf16_rn(tile[c0 + 3][cr]);
    *(ushort4*)(dst + base + (size_t)(bx + cr) * R + by + c0) = o;
}

// ---------------------------------------------------------------------------
// Stage 2: router. One wave (32 lanes) per token. Also converts x -> bf16.
// ---------------------------------------------------------------------------
__global__ __launch_bounds__(256) void router_kernel(
    const float* __restrict__ x, const float* __restrict__ Wr,
    const float* __restrict__ br, unsigned short* __restrict__ xbf,
    int* __restrict__ cnt, int* __restrict__ list, float* __restrict__ gw) {

    int lane = threadIdx.x & 31;
    int wave = threadIdx.x >> 5;
    int token = blockIdx.x * 8 + wave;   // grid = NT/8 blocks

    const float* xr = x + (size_t)token * DIM;
    float acc[NE];
    #pragma unroll
    for (int e = 0; e < NE; ++e) acc[e] = 0.0f;

    for (int d = lane; d < DIM; d += 32) {
        float xv = xr[d];
        xbf[(size_t)token * DIM + d] = f32_to_bf16_rn(xv);
        const float4* w4 = (const float4*)(Wr + (size_t)d * NE);
        float4 wa = w4[0], wb = w4[1];
        acc[0] += xv * wa.x; acc[1] += xv * wa.y;
        acc[2] += xv * wa.z; acc[3] += xv * wa.w;
        acc[4] += xv * wb.x; acc[5] += xv * wb.y;
        acc[6] += xv * wb.z; acc[7] += xv * wb.w;
    }
    #pragma unroll
    for (int e = 0; e < NE; ++e) {
        #pragma unroll
        for (int off = 16; off > 0; off >>= 1)
            acc[e] += __shfl_xor(acc[e], off, 32);
    }
    if (lane == 0) {
        float lg[NE];
        #pragma unroll
        for (int e = 0; e < NE; ++e) lg[e] = acc[e] + br[e];
        int i0 = 0;
        #pragma unroll
        for (int e = 1; e < NE; ++e) if (lg[e] > lg[i0]) i0 = e;
        int i1 = (i0 == 0) ? 1 : 0;
        #pragma unroll
        for (int e = 0; e < NE; ++e) if (e != i0 && lg[e] > lg[i1]) i1 = e;
        float p0 = 1.0f / (1.0f + __expf(lg[i1] - lg[i0]));
        float p1 = 1.0f - p0;
        int s0 = atomicAdd(&cnt[i0], 1);
        list[i0 * NT + s0] = token;  gw[i0 * NT + s0] = p0;
        int s1 = atomicAdd(&cnt[i1], 1);
        list[i1 * NT + s1] = token;  gw[i1 * NT + s1] = p1;
    }
}

// ---------------------------------------------------------------------------
// Stage 3: fused grouped FFN.
//   block = 256 threads = 8 waves; tile = 32 tokens of one expert.
//   GEMM1: K staged 128 deep -> 4 WMMAs per barrier round.
//   GEMM2: 16 WMMAs per wave per staging round, 32x1024 f32 acc in VGPRs.
//   All staging is contiguous 16B copies via GLOBAL_LOAD_ASYNC_TO_LDS_B128.
// LDS: xs[32][1024] | w1s[64f][128d] | w2s[1024n][32k] | hs[32][64] | tok/gate
// ---------------------------------------------------------------------------
#define SM_XS   0                        // 32*1024*2 = 65536
#define SM_W1S  65536                    // 64*128*2  = 16384
#define SM_W2S  81920                    // 1024*32*2 = 65536
#define SM_HS   147456                   // 32*64*2   = 4096
#define SM_TOK  151552                   // 32*4
#define SM_GATE 151680                   // 32*4
#define SM_SIZE 151808

__global__ __launch_bounds__(256, 1) void ffn_kernel(
    const unsigned short* __restrict__ xbf,
    const unsigned short* __restrict__ w1t,   // [E][DFF][D]  (B1 = W1^T rows)
    const unsigned short* __restrict__ w2t,   // [E][D][DFF]  (B2 = W2^T rows)
    const float* __restrict__ b1, const float* __restrict__ b2,
    const int* __restrict__ cnt, const int* __restrict__ list,
    const float* __restrict__ gw, float* __restrict__ out) {

    extern __shared__ char smem[];
    unsigned short* xs  = (unsigned short*)(smem + SM_XS);
    unsigned short* w1s = (unsigned short*)(smem + SM_W1S);
    unsigned short* w2s = (unsigned short*)(smem + SM_W2S);
    unsigned short* hs  = (unsigned short*)(smem + SM_HS);
    int*   toks  = (int*)(smem + SM_TOK);
    float* gates = (float*)(smem + SM_GATE);

    int e    = blockIdx.x / TILES_PER_E;
    int tile = blockIdx.x % TILES_PER_E;
    int count = cnt[e];
    if (tile * MT >= count) return;      // block-uniform: EXEC stays all-1s

    int t = threadIdx.x;
    int lane = t & 31, wave = t >> 5;
    int wm = wave & 1;                   // M subtile (0..1)
    int wq = wave >> 1;                  // f-subtile (GEMM1) / N-quadrant (GEMM2)
    int row = lane & 15;
    int kh  = lane >> 4;

    if (t < MT) {
        int idx = tile * MT + t;
        if (idx < count) { toks[t] = list[e * NT + idx]; gates[t] = gw[e * NT + idx]; }
        else             { toks[t] = -1;                 gates[t] = 0.0f; }
    }
    __syncthreads();

    // ---- stage x tile: 32 rows x 1024 bf16 (gather by token id, async) ----
    {
        int m = t >> 3, seg = t & 7;     // 8 threads per row, 128 bf16 each
        int tok = toks[m];
        unsigned short* dst = xs + m * DIM + seg * 128;
        if (tok >= 0) {
            const unsigned short* src = xbf + (size_t)tok * DIM + seg * 128;
            #pragma unroll
            for (int i = 0; i < 16; ++i) copy16_g2lds(src + i * 8, dst + i * 8);
        } else {
            uint4 z = {0u, 0u, 0u, 0u};
            #pragma unroll
            for (int i = 0; i < 16; ++i) *(uint4*)(dst + i * 8) = z;
        }
    }
    async_fence();
    __syncthreads();

    const unsigned short* w1te = w1t + (size_t)e * DFF_ * DIM;
    const unsigned short* w2te = w2t + (size_t)e * DIM * DFF_;

    f32x8 acc2[16];
    #pragma unroll
    for (int j = 0; j < 16; ++j)
        #pragma unroll
        for (int r = 0; r < 8; ++r) acc2[j][r] = 0.0f;

    for (int kc = 0; kc < DFF_; kc += 64) {          // DFF chunk
        if (kc + 64 < DFF_)
            __builtin_prefetch(w1te + (size_t)(kc + 64) * DIM, 0, 0);

        // ---------- GEMM1: h[32x64] = x[32x1024] @ W1[:, kc:kc+64] --------
        f32x8 acc1;
        #pragma unroll
        for (int r = 0; r < 8; ++r) acc1[r] = 0.0f;

        for (int kd = 0; kd < DIM; kd += 128) {
            // stage w1s[f][d]: 64 rows (f) x 128 (d), contiguous from W1T
            {
                int f   = t >> 2;          // 0..63
                int seg = (t & 3) * 32;    // 0..96 (bf16)
                const unsigned short* src = w1te + (size_t)(kc + f) * DIM + kd + seg;
                unsigned short*       dst = w1s + f * 128 + seg;
                #pragma unroll
                for (int i = 0; i < 4; ++i) copy16_g2lds(src + i * 8, dst + i * 8);
            }
            async_fence();
            __syncthreads();

            #pragma unroll
            for (int kk = 0; kk < 128; kk += 32) {
                FragU a;   // A 16x32 bf16: VGPR v -> K = (v&4?16:0)+8*kh+2*(v&3)
                #pragma unroll
                for (int v = 0; v < 8; ++v) {
                    int k = ((v & 4) ? 16 : 0) + 8 * kh + 2 * (v & 3);
                    int base = (wm * 16 + row) * DIM + kd + kk + k;
                    a.u[2 * v]     = xs[base];
                    a.u[2 * v + 1] = xs[base + 1];
                }
                FragU b;   // B 32x16 bf16: VGPR v -> K = (v&4?16:0)+8*kh+2*(v&3)
                #pragma unroll
                for (int v = 0; v < 8; ++v) {
                    int base = (wq * 16 + row) * 128 + kk + ((v & 4) ? 16 : 0) + 8 * kh + 2 * (v & 3);
                    b.u[2 * v]     = w1s[base];
                    b.u[2 * v + 1] = w1s[base + 1];
                }
                acc1 = __builtin_amdgcn_wmma_f32_16x16x32_bf16(
                    false, a.v, false, b.v, (short)0, acc1, false, false);
            }
            __syncthreads();
        }

        // bias + gelu -> hs (C layout: VGPR r -> M = r + 8*kh, N = row)
        {
            float bb = b1[e * DFF_ + kc + wq * 16 + row];
            #pragma unroll
            for (int r = 0; r < 8; ++r) {
                float v = gelu_tanh(acc1[r] + bb);
                int mrow = wm * 16 + r + 8 * kh;
                hs[mrow * 64 + wq * 16 + row] = f32_to_bf16_rn(v);
            }
        }
        __syncthreads();

        // ---------- GEMM2: y[32x1024] += h[32x64] @ W2[kc:kc+64, :] -------
        for (int kf = 0; kf < 64; kf += 32) {
            // stage w2s[n][k]: 1024 rows (n) x 32 (k), contiguous from W2T
            {
                int n0 = t * 4;
                #pragma unroll
                for (int r = 0; r < 4; ++r) {
                    const unsigned short* src =
                        w2te + (size_t)(n0 + r) * DFF_ + kc + kf;
                    unsigned short* dst = w2s + (n0 + r) * 32;
                    #pragma unroll
                    for (int c = 0; c < 4; ++c)
                        copy16_g2lds(src + c * 8, dst + c * 8);
                }
            }
            async_fence();
            __syncthreads();

            FragU a2;
            #pragma unroll
            for (int v = 0; v < 8; ++v) {
                int k = ((v & 4) ? 16 : 0) + 8 * kh + 2 * (v & 3);
                int base = (wm * 16 + row) * 64 + kf + k;
                a2.u[2 * v]     = hs[base];
                a2.u[2 * v + 1] = hs[base + 1];
            }
            #pragma unroll
            for (int j = 0; j < 16; ++j) {
                FragU bf;
                int n = wq * 256 + j * 16 + row;
                #pragma unroll
                for (int v = 0; v < 8; ++v) {
                    int base = n * 32 + ((v & 4) ? 16 : 0) + 8 * kh + 2 * (v & 3);
                    bf.u[2 * v]     = w2s[base];
                    bf.u[2 * v + 1] = w2s[base + 1];
                }
                acc2[j] = __builtin_amdgcn_wmma_f32_16x16x32_bf16(
                    false, a2.v, false, bf.v, (short)0, acc2[j], false, false);
            }
            __syncthreads();
        }
    }

    // ---- epilogue: out[token] += gate * (y + b2[e]) ----
    #pragma unroll
    for (int r = 0; r < 8; ++r) {
        int mrow = wm * 16 + r + 8 * kh;
        int token = toks[mrow];
        if (token < 0) continue;
        float gate = gates[mrow];
        #pragma unroll
        for (int j = 0; j < 16; ++j) {
            int n = wq * 256 + j * 16 + row;
            float val = gate * (acc2[j][r] + b2[e * DIM + n]);
            atomicAdd(out + (size_t)token * DIM + n, val);
        }
    }
}

// ---------------------------------------------------------------------------
// Host launcher
// ---------------------------------------------------------------------------
extern "C" void kernel_launch(void* const* d_in, const int* in_sizes, int n_in,
                              void* d_out, int out_size, void* d_ws, size_t ws_size,
                              hipStream_t stream) {
    const float* x  = (const float*)d_in[0];   // [B,S,D]
    const float* Wr = (const float*)d_in[1];   // [D,E]
    const float* br = (const float*)d_in[2];   // [E]
    const float* W1 = (const float*)d_in[3];   // [E,D,DFF]
    const float* b1 = (const float*)d_in[4];   // [E,DFF]
    const float* W2 = (const float*)d_in[5];   // [E,DFF,D]
    const float* b2 = (const float*)d_in[6];   // [E,D]
    float* out = (float*)d_out;

    // workspace carve (aligned offsets)
    char* ws = (char*)d_ws;
    int*            cnt  = (int*)(ws + 0);                 //       32 B
    int*            list = (int*)(ws + 1024);              //  256 KiB
    float*          gw   = (float*)(ws + 1024 + 262144);   //  256 KiB
    unsigned short* xbf  = (unsigned short*)(ws + 525312);           // 16 MiB
    unsigned short* w1t  = (unsigned short*)(ws + 525312 + 16777216);// 64 MiB
    unsigned short* w2t  = (unsigned short*)(ws + 525312 + 16777216 + 67108864);

    (void)hipMemsetAsync(cnt, 0, NE * sizeof(int), stream);
    (void)hipMemsetAsync(out, 0, (size_t)out_size * sizeof(float), stream);

    // W1 [E][D][DFF] -> W1T [E][DFF][D];  W2 [E][DFF][D] -> W2T [E][D][DFF]
    transpose_cvt_kernel<<<dim3(DFF_ / 32, DIM / 32, NE), 256, 0, stream>>>(
        W1, w1t, DIM, DFF_);
    transpose_cvt_kernel<<<dim3(DIM / 32, DFF_ / 32, NE), 256, 0, stream>>>(
        W2, w2t, DFF_, DIM);

    router_kernel<<<NT / 8, 256, 0, stream>>>(x, Wr, br, xbf, cnt, list, gw);

    ffn_kernel<<<NE * TILES_PER_E, 256, SM_SIZE, stream>>>(
        xbf, w1t, w2t, b1, b2, cnt, list, gw, out);
}
